// sLSTM_23708219474074
// MI455X (gfx1250) — compile-verified, matrix operand order checked by reading, not run on hardware
//
#include <hip/hip_runtime.h>
#include <hip/hip_bf16.h>
#include <math.h>
#include <stdint.h>

typedef __attribute__((ext_vector_type(16))) __bf16 v16bf;
typedef __attribute__((ext_vector_type(8)))  __bf16 bf16x8;
typedef __attribute__((ext_vector_type(8)))  float  v8f;

#define LDSS 40                      // LDS row stride in bf16 (32 data + 8 pad)
#define TILE_ELEMS (128 * LDSS)      // one 128x32 tile (padded) in bf16 elems
#define SZ_BD ((size_t)16384 * 1024)

// ---------------------------------------------------------------------------
// Async tile loader: 128 rows x 32 bf16 from global directly into LDS via
// GLOBAL_LOAD_ASYNC_TO_LDS_B128 (CDNA5 async path, tracked by ASYNCcnt).
// 512 chunks of 8 bf16 (16B); 2 chunks per thread, straight-line.
// Invalid (padded) rows are zero-filled synchronously via ds_store.
// ---------------------------------------------------------------------------
static __device__ __forceinline__ void async_load_tile(const __bf16* __restrict__ g,
                                                       long ldg, int k0, int nvalid,
                                                       __bf16* sh, int tid) {
#pragma unroll
  for (int u = 0; u < 2; ++u) {
    int c  = tid + u * 256;
    int r  = c >> 2;
    int kp = (c & 3) * 8;
    __bf16* sp = sh + r * LDSS + kp;
    if (r < nvalid) {
      const __bf16* p = g + (long)r * ldg + k0 + kp;
      unsigned lds_off = (unsigned)(unsigned long long)(uintptr_t)sp; // low 32b = LDS offset
      asm volatile("global_load_async_to_lds_b128 %0, %1, off"
                   :
                   : "v"(lds_off), "v"(p)
                   : "memory");
    } else {
      bf16x8 z;
#pragma unroll
      for (int i = 0; i < 8; ++i) z[i] = (__bf16)0.0f;
      *(bf16x8*)sp = z;
    }
  }
}

static __device__ __forceinline__ void wait_async_lds() {
  asm volatile("s_wait_asynccnt 0x0" ::: "memory");
}

// ---------------------------------------------------------------------------
// One 32-deep K-step of WMMA for a wave's 32x64 sub-tile (2x4 of 16x16).
// A frag (16x32 bf16): lane 0-15 -> M=lane, K 0..7 & 16..23; lane 16-31 -> K 8..15 & 24..31.
// B frag (32x16 bf16): same K chunking, lane%16 selects column N.
// ---------------------------------------------------------------------------
static __device__ __forceinline__ void wave_mma(const __bf16* Ash, const __bf16* Bsh,
                                                int wm, int wn, int lane, v8f acc[][4]) {
  int l16 = lane & 15;
  int kb  = (lane >> 4) * 8;
  v16bf a[2], b[4];
#pragma unroll
  for (int mi = 0; mi < 2; ++mi) {
    const __bf16* p = Ash + (wm * 32 + mi * 16 + l16) * LDSS + kb;
    bf16x8 lo = *(const bf16x8*)p;
    bf16x8 hi = *(const bf16x8*)(p + 16);
    a[mi] = __builtin_shufflevector(lo, hi, 0,1,2,3,4,5,6,7,8,9,10,11,12,13,14,15);
  }
#pragma unroll
  for (int ni = 0; ni < 4; ++ni) {
    const __bf16* p = Bsh + (wn * 64 + ni * 16 + l16) * LDSS + kb;
    bf16x8 lo = *(const bf16x8*)p;
    bf16x8 hi = *(const bf16x8*)(p + 16);
    b[ni] = __builtin_shufflevector(lo, hi, 0,1,2,3,4,5,6,7,8,9,10,11,12,13,14,15);
  }
#pragma unroll
  for (int mi = 0; mi < 2; ++mi)
#pragma unroll
    for (int ni = 0; ni < 4; ++ni)
      acc[mi][ni] = __builtin_amdgcn_wmma_f32_16x16x32_bf16(
          false, a[mi], false, b[ni], (short)0, acc[mi][ni], false, false);
}

static __device__ __forceinline__ void zero_acc(v8f acc[][4]) {
#pragma unroll
  for (int mi = 0; mi < 2; ++mi)
#pragma unroll
    for (int ni = 0; ni < 4; ++ni)
#pragma unroll
      for (int r = 0; r < 8; ++r) acc[mi][ni][r] = 0.0f;
}

// ---------------------------------------------------------------------------
// Double-buffered K-loop over a 128(M) x 128(N) x K slice.  Async loads for
// tile s+1 are issued before the WMMAs of tile s, so memory latency hides
// behind matrix work.  Ash/Bsh each hold TWO tiles (2*TILE_ELEMS).
// ---------------------------------------------------------------------------
static __device__ __forceinline__ void mm_loop(const __bf16* Ag, long lda,
                                               const __bf16* Bg, long ldb, int nvalid,
                                               int K, __bf16* Ash, __bf16* Bsh,
                                               int tid, int wm, int wn, int lane,
                                               v8f acc[][4]) {
  const int nsteps = K / 32;
  async_load_tile(Ag, lda, 0, 1 << 30, Ash, tid);
  async_load_tile(Bg, ldb, 0, nvalid,  Bsh, tid);
  wait_async_lds();
  __syncthreads();
  for (int s = 0; s < nsteps; ++s) {
    int cur = (s & 1) * TILE_ELEMS;
    int nxt = ((s + 1) & 1) * TILE_ELEMS;
    if (s + 1 < nsteps) {
      async_load_tile(Ag, lda, (s + 1) * 32, 1 << 30, Ash + nxt, tid);
      async_load_tile(Bg, ldb, (s + 1) * 32, nvalid,  Bsh + nxt, tid);
    }
    if (s + 2 < nsteps) {   // light prefetch two steps ahead
      __builtin_prefetch(Ag + (long)(tid >> 2) * lda + (s + 2) * 32, 0, 1);
      __builtin_prefetch(Bg + (long)(tid >> 2) * ldb + (s + 2) * 32, 0, 1);
    }
    wave_mma(Ash + cur, Bsh + cur, wm, wn, lane, acc);
    wait_async_lds();   // tile s+1 has landed in LDS (this wave's issues)
    __syncthreads();    // drains DScnt first -> safe to overwrite cur next iter
  }
}

// ---------------------------------------------------------------------------
// fp32 -> bf16 converters
// ---------------------------------------------------------------------------
__global__ __launch_bounds__(256) void cvt_bf16(const float* __restrict__ s,
                                                __bf16* __restrict__ d, size_t n) {
  size_t i = (size_t)blockIdx.x * 256 + threadIdx.x;
  if (i < n) d[i] = (__bf16)s[i];
}

__global__ __launch_bounds__(256) void cvt_pad_bf16(const float* __restrict__ s,
                                                    __bf16* __restrict__ d,
                                                    int cols, int padcols, size_t total) {
  size_t i = (size_t)blockIdx.x * 256 + threadIdx.x;
  if (i < total) {
    size_t r = i / padcols;
    int    c = (int)(i % padcols);
    d[i] = (c < cols) ? (__bf16)s[r * cols + c] : (__bf16)0.0f;
  }
}

// ---------------------------------------------------------------------------
// LayerNorm(seq) -> bf16 x.  One 256-thread block per row, float4 per thread.
// ---------------------------------------------------------------------------
__global__ __launch_bounds__(256) void ln_cast(const float* __restrict__ seq,
                                               const float* __restrict__ w,
                                               const float* __restrict__ b,
                                               __bf16* __restrict__ xbf) {
  __shared__ float sm[8];
  int row = blockIdx.x, tid = threadIdx.x;
  int lane = tid & 31, wid = tid >> 5;
  float4 v = *(const float4*)(seq + (size_t)row * 1024 + tid * 4);
  float s = v.x + v.y + v.z + v.w;
#pragma unroll
  for (int o = 16; o > 0; o >>= 1) s += __shfl_xor(s, o, 32);
  if (lane == 0) sm[wid] = s;
  __syncthreads();
  float tot = 0.f;
#pragma unroll
  for (int i = 0; i < 8; ++i) tot += sm[i];
  float mu = tot * (1.0f / 1024.0f);
  float d0 = v.x - mu, d1 = v.y - mu, d2 = v.z - mu, d3 = v.w - mu;
  float q = d0 * d0 + d1 * d1 + d2 * d2 + d3 * d3;
#pragma unroll
  for (int o = 16; o > 0; o >>= 1) q += __shfl_xor(q, o, 32);
  __syncthreads();
  if (lane == 0) sm[wid] = q;
  __syncthreads();
  float qt = 0.f;
#pragma unroll
  for (int i = 0; i < 8; ++i) qt += sm[i];
  float inv = rsqrtf(qt * (1.0f / 1024.0f) + 1e-5f);
  int c = tid * 4;
  __bf16* outp = xbf + (size_t)row * 1024 + c;
  outp[0] = (__bf16)(d0 * inv * w[c + 0] + b[c + 0]);
  outp[1] = (__bf16)(d1 * inv * w[c + 1] + b[c + 1]);
  outp[2] = (__bf16)(d2 * inv * w[c + 2] + b[c + 2]);
  outp[3] = (__bf16)(d3 * inv * w[c + 3] + b[c + 3]);
}

// ---------------------------------------------------------------------------
// Gate GEMM: P[g] = x @ Wg^T + bg + blockdiag(h0, Rg).
// grid = (Mtiles=128, heads=8, gates=4), 256 threads (8 waves).
// N tile (128) == one head, so the recurrence folds into the same accumulators.
// ---------------------------------------------------------------------------
__global__ __launch_bounds__(256) void gemm_gates(
    const __bf16* __restrict__ xbf, const __bf16* __restrict__ h0bf,
    const __bf16* Wi, const __bf16* Wf, const __bf16* Wz, const __bf16* Wo,
    const __bf16* Ri, const __bf16* Rf, const __bf16* Rz, const __bf16* Ro,
    const float* bi, const float* bfg, const float* bz, const float* bo,
    float* __restrict__ P) {
  __shared__ __bf16 Ash[2 * TILE_ELEMS];
  __shared__ __bf16 Bsh[2 * TILE_ELEMS];
  int tid = threadIdx.x;
  int mbase = blockIdx.x * 128;
  int head  = blockIdx.y;
  int g     = blockIdx.z;
  const __bf16* Wg = (g == 0) ? Wi : (g == 1) ? Wf : (g == 2) ? Wz : Wo;
  const __bf16* Rg = (g == 0) ? Ri : (g == 1) ? Rf : (g == 2) ? Rz : Ro;
  const float*  bg = (g == 0) ? bi : (g == 1) ? bfg : (g == 2) ? bz : bo;
  int nbase = head * 128;
  int wid = tid >> 5, lane = tid & 31, wm = wid & 3, wn = wid >> 2;
  v8f acc[2][4];
  zero_acc(acc);
  // x @ Wg^T over K=1024
  mm_loop(xbf + (size_t)mbase * 1024, 1024,
          Wg + (size_t)nbase * 1024, 1024, 1 << 30,
          1024, Ash, Bsh, tid, wm, wn, lane, acc);
  // + h0[:, head] @ Rg[head]^T over K=128 (block-diagonal recurrence)
  mm_loop(h0bf + (size_t)mbase * 1024 + nbase, 1024,
          Rg + (size_t)head * 128 * 128, 128, 1 << 30,
          128, Ash, Bsh, tid, wm, wn, lane, acc);
  float* Pg = P + (size_t)g * SZ_BD;
  int l16 = lane & 15, mh = (lane >> 4) * 8;
#pragma unroll
  for (int mi = 0; mi < 2; ++mi)
#pragma unroll
    for (int ni = 0; ni < 4; ++ni)
#pragma unroll
      for (int r = 0; r < 8; ++r) {
        int M = mbase + wm * 32 + mi * 16 + mh + r;
        int N = nbase + wn * 64 + ni * 16 + l16;
        Pg[(size_t)M * 1024 + N] = acc[mi][ni][r] + bg[N];
      }
}

// ---------------------------------------------------------------------------
// sLSTM state update + GroupNorm, fused. One block per batch row; one wave
// per head (128 channels, 4 per lane) -> shfl reduction for GN stats.
// ---------------------------------------------------------------------------
__global__ __launch_bounds__(256) void state_pointwise(
    const float* __restrict__ P, const float* __restrict__ c0,
    const float* __restrict__ n0, const float* __restrict__ m0,
    const float* __restrict__ gnw, const float* __restrict__ gnb,
    float* __restrict__ dout, __bf16* __restrict__ gnbf) {
  const size_t S = SZ_BD;
  int b = blockIdx.x, tid = threadIdx.x;
  int lane = tid & 31, wid = tid >> 5;
  float hl[4]; int chs[4];
  float s = 0.f, ss = 0.f;
#pragma unroll
  for (int j = 0; j < 4; ++j) {
    int ch = wid * 128 + j * 32 + lane;
    size_t idx = (size_t)b * 1024 + ch;
    float iv = P[idx], fv = P[S + idx], zv = P[2 * S + idx], ov = P[3 * S + idx];
    float m0v = m0[idx], c0v = c0[idx], n0v = n0[idx];
    float mt = fmaxf(fv + m0v, iv);
    float ie = expf(iv - mt);
    float fe = expf(fv - mt + m0v);
    float za = tanhf(zv);
    float oa = 1.0f / (1.0f + expf(-ov));
    float ct = fe * c0v + ie * za;
    float nt = fe * n0v + ie;
    float ht = oa * (ct / nt);
    dout[S + idx]     = ct;
    dout[2 * S + idx] = nt;
    dout[3 * S + idx] = ht;
    dout[4 * S + idx] = mt;
    hl[j] = ht; chs[j] = ch;
    s += ht; ss += ht * ht;
  }
#pragma unroll
  for (int o = 16; o > 0; o >>= 1) { s += __shfl_xor(s, o, 32); ss += __shfl_xor(ss, o, 32); }
  float mu  = s  * (1.0f / 128.0f);
  float var = ss * (1.0f / 128.0f) - mu * mu;
  float inv = rsqrtf(var + 1e-5f);
#pragma unroll
  for (int j = 0; j < 4; ++j) {
    int ch = chs[j];
    float gv = (hl[j] - mu) * inv * gnw[ch] + gnb[ch];
    gnbf[(size_t)b * 1024 + ch] = (__bf16)gv;
  }
}

// ---------------------------------------------------------------------------
// Up projection: U = gn @ up_w^T + up_b  (N=2730, padded tiles to 2816)
// ---------------------------------------------------------------------------
__global__ __launch_bounds__(256) void gemm_up(const __bf16* __restrict__ gnbf,
                                               const __bf16* __restrict__ upw,
                                               const float* __restrict__ upb,
                                               float* __restrict__ U) {
  __shared__ __bf16 Ash[2 * TILE_ELEMS];
  __shared__ __bf16 Bsh[2 * TILE_ELEMS];
  int tid = threadIdx.x;
  int mbase = blockIdx.x * 128, nbase = blockIdx.y * 128;
  int wid = tid >> 5, lane = tid & 31, wm = wid & 3, wn = wid >> 2;
  v8f acc[2][4];
  zero_acc(acc);
  mm_loop(gnbf + (size_t)mbase * 1024, 1024,
          upw + (size_t)nbase * 1024, 1024, 2730 - nbase,
          1024, Ash, Bsh, tid, wm, wn, lane, acc);
  int l16 = lane & 15, mh = (lane >> 4) * 8;
#pragma unroll
  for (int mi = 0; mi < 2; ++mi)
#pragma unroll
    for (int ni = 0; ni < 4; ++ni)
#pragma unroll
      for (int r = 0; r < 8; ++r) {
        int M = mbase + wm * 32 + mi * 16 + mh + r;
        int N = nbase + wn * 64 + ni * 16 + l16;
        if (N < 2730) U[(size_t)M * 2730 + N] = acc[mi][ni][r] + upb[N];
      }
}

// GLU with exact GELU; writes bf16 activation padded to 1408 cols (zeros).
__global__ __launch_bounds__(256) void glu_combine(const float* __restrict__ U,
                                                   __bf16* __restrict__ act) {
  size_t idx = (size_t)blockIdx.x * 256 + threadIdx.x;   // over 16384*1408
  size_t row = idx / 1408;
  int    col = (int)(idx % 1408);
  __bf16 r = (__bf16)0.0f;
  if (col < 1365) {
    float v1 = U[row * 2730 + col];
    float v2 = U[row * 2730 + 1365 + col];
    float g  = 0.5f * v2 * (1.0f + erff(v2 * 0.70710678118654752f));
    r = (__bf16)(v1 + g);
  }
  act[idx] = r;
}

// ---------------------------------------------------------------------------
// Down projection + bias + residual: out = act @ down_w^T + down_b + seq
// K padded 1365 -> 1408 (both operands zero-padded).
// ---------------------------------------------------------------------------
__global__ __launch_bounds__(256) void gemm_down(const __bf16* __restrict__ actbf,
                                                 const __bf16* __restrict__ dwb,
                                                 const float* __restrict__ db,
                                                 const float* __restrict__ seq,
                                                 float* __restrict__ dout) {
  __shared__ __bf16 Ash[2 * TILE_ELEMS];
  __shared__ __bf16 Bsh[2 * TILE_ELEMS];
  int tid = threadIdx.x;
  int mbase = blockIdx.x * 128, nbase = blockIdx.y * 128;
  int wid = tid >> 5, lane = tid & 31, wm = wid & 3, wn = wid >> 2;
  v8f acc[2][4];
  zero_acc(acc);
  mm_loop(actbf + (size_t)mbase * 1408, 1408,
          dwb + (size_t)nbase * 1408, 1408, 1 << 30,
          1408, Ash, Bsh, tid, wm, wn, lane, acc);
  int l16 = lane & 15, mh = (lane >> 4) * 8;
#pragma unroll
  for (int mi = 0; mi < 2; ++mi)
#pragma unroll
    for (int ni = 0; ni < 4; ++ni)
#pragma unroll
      for (int r = 0; r < 8; ++r) {
        int M = mbase + wm * 32 + mi * 16 + mh + r;
        int N = nbase + wn * 64 + ni * 16 + l16;
        size_t idx = (size_t)M * 1024 + N;
        dout[idx] = acc[mi][ni][r] + db[N] + seq[idx];
      }
}

// ---------------------------------------------------------------------------
extern "C" void kernel_launch(void* const* d_in, const int* in_sizes, int n_in,
                              void* d_out, int out_size, void* d_ws, size_t ws_size,
                              hipStream_t stream) {
  (void)in_sizes; (void)n_in; (void)out_size; (void)ws_size;
  const float* seq    = (const float*)d_in[0];
  const float* c0     = (const float*)d_in[1];
  const float* n0     = (const float*)d_in[2];
  const float* h0     = (const float*)d_in[3];
  const float* m0     = (const float*)d_in[4];
  const float* ln_w   = (const float*)d_in[5];
  const float* ln_b   = (const float*)d_in[6];
  const float* Wz     = (const float*)d_in[7];
  const float* bz     = (const float*)d_in[8];
  const float* Wi     = (const float*)d_in[9];
  const float* bi     = (const float*)d_in[10];
  const float* Wo     = (const float*)d_in[11];
  const float* bo     = (const float*)d_in[12];
  const float* Wf     = (const float*)d_in[13];
  const float* bfv    = (const float*)d_in[14];
  const float* Rz     = (const float*)d_in[15];
  const float* Ri     = (const float*)d_in[16];
  const float* Ro     = (const float*)d_in[17];
  const float* Rf     = (const float*)d_in[18];
  const float* gn_w   = (const float*)d_in[19];
  const float* gn_b   = (const float*)d_in[20];
  const float* up_w   = (const float*)d_in[21];
  const float* up_b   = (const float*)d_in[22];
  const float* down_w = (const float*)d_in[23];
  const float* down_b = (const float*)d_in[24];
  float* dout = (float*)d_out;

  unsigned char* WS = (unsigned char*)d_ws;
  size_t off = 0;
  auto alloc = [&](size_t bytes) { size_t o = off; off += (bytes + 255) & ~(size_t)255; return o; };
  __bf16* xbf   = (__bf16*)(WS + alloc(SZ_BD * 2));
  __bf16* h0bf  = (__bf16*)(WS + alloc(SZ_BD * 2));
  __bf16* Wib   = (__bf16*)(WS + alloc((size_t)1024 * 1024 * 2));
  __bf16* Wfb   = (__bf16*)(WS + alloc((size_t)1024 * 1024 * 2));
  __bf16* Wzb   = (__bf16*)(WS + alloc((size_t)1024 * 1024 * 2));
  __bf16* Wob   = (__bf16*)(WS + alloc((size_t)1024 * 1024 * 2));
  __bf16* Rib   = (__bf16*)(WS + alloc((size_t)131072 * 2));
  __bf16* Rfb   = (__bf16*)(WS + alloc((size_t)131072 * 2));
  __bf16* Rzb   = (__bf16*)(WS + alloc((size_t)131072 * 2));
  __bf16* Rob   = (__bf16*)(WS + alloc((size_t)131072 * 2));
  __bf16* upwb  = (__bf16*)(WS + alloc((size_t)2730 * 1024 * 2));
  __bf16* dwb   = (__bf16*)(WS + alloc((size_t)1024 * 1408 * 2));
  __bf16* gnbf  = (__bf16*)(WS + alloc(SZ_BD * 2));
  __bf16* actbf = (__bf16*)(WS + alloc((size_t)16384 * 1408 * 2));
  // P (4 x B x 1024 fp32, 256MB) and U (B x 2730 fp32, ~171MB) are live in
  // disjoint phases -> share the same region.
  size_t big = off;
  float* P = (float*)(WS + big);
  float* U = (float*)(WS + big);

  auto blks = [](size_t n) { return dim3((unsigned)((n + 255) / 256)); };

  // 1) bf16 conversions of states/weights
  cvt_bf16<<<blks(SZ_BD), 256, 0, stream>>>(h0, h0bf, SZ_BD);
  cvt_bf16<<<blks(1048576), 256, 0, stream>>>(Wi, Wib, 1048576);
  cvt_bf16<<<blks(1048576), 256, 0, stream>>>(Wf, Wfb, 1048576);
  cvt_bf16<<<blks(1048576), 256, 0, stream>>>(Wz, Wzb, 1048576);
  cvt_bf16<<<blks(1048576), 256, 0, stream>>>(Wo, Wob, 1048576);
  cvt_bf16<<<blks(131072), 256, 0, stream>>>(Ri, Rib, 131072);
  cvt_bf16<<<blks(131072), 256, 0, stream>>>(Rf, Rfb, 131072);
  cvt_bf16<<<blks(131072), 256, 0, stream>>>(Rz, Rzb, 131072);
  cvt_bf16<<<blks(131072), 256, 0, stream>>>(Ro, Rob, 131072);
  cvt_bf16<<<blks((size_t)2730 * 1024), 256, 0, stream>>>(up_w, upwb, (size_t)2730 * 1024);
  cvt_pad_bf16<<<blks((size_t)1024 * 1408), 256, 0, stream>>>(down_w, dwb, 1365, 1408,
                                                              (size_t)1024 * 1408);
  // 2) LayerNorm + cast
  ln_cast<<<16384, 256, 0, stream>>>(seq, ln_w, ln_b, xbf);
  // 3) gate GEMMs (+ block-diagonal recurrence)
  gemm_gates<<<dim3(128, 8, 4), 256, 0, stream>>>(xbf, h0bf,
                                                  Wib, Wfb, Wzb, Wob,
                                                  Rib, Rfb, Rzb, Rob,
                                                  bi, bfv, bz, bo, P);
  // 4) sLSTM state update + GroupNorm (writes c,n,h,m to d_out; gn -> bf16)
  state_pointwise<<<16384, 256, 0, stream>>>(P, c0, n0, m0, gn_w, gn_b, dout, gnbf);
  // 5) up projection
  gemm_up<<<dim3(128, 22), 256, 0, stream>>>(gnbf, upwb, up_b, U);
  // 6) GLU + GELU -> padded bf16 activation
  glu_combine<<<blks((size_t)16384 * 1408), 256, 0, stream>>>(U, actbf);
  // 7) down projection + bias + residual
  gemm_down<<<dim3(128, 8), 256, 0, stream>>>(actbf, dwb, down_b, seq, dout);
}
